// SelfAttention_22308060136280
// MI455X (gfx1250) — matrix operand order repaired and checked by reading, not compiled, not run on hardware
//
#include <hip/hip_runtime.h>
#include <hip/hip_bf16.h>
#include <math.h>

// ---------------------------------------------------------------------------
// SelfAttention fused kernel for MI455X (gfx1250, wave32, WMMA)
// B=512, N=H=256.  out[b,m] = (sum_k s[k]*e[k,m]) / (sum_k e[k,m])
// where e[k,m]=exp(scores[k,m]-colmax[m]), s[k]=sum_n feats[b,n,k].
// ---------------------------------------------------------------------------

typedef __attribute__((ext_vector_type(16))) _Float16 v16h;
typedef __attribute__((ext_vector_type(8)))  _Float16 v8h;
typedef __attribute__((ext_vector_type(4)))  _Float16 v4h;
typedef __attribute__((ext_vector_type(8)))  float    v8f;

#define NTOK 256      // N == H == 256
#define NTHREADS 256  // 8 waves of 32

// LDS layout (bytes)
#define OFF_FEATS   0                 // 256*256 f16 = 131072
#define OFF_A       131072            // 256*256 f16 = 131072
#define OFF_SCORES  262144            // 256*32 f32  =  32768 (also phase-A scratch)
#define OFF_S       294912            // 256 f32
#define OFF_BIAS    295936            // 256 f32
#define OFF_RED     296960            // 512 f32
#define OFF_CMAX    299008            // 32 f32
#define SMEM_BYTES  299136

__device__ __forceinline__ v8f wmma_f16(v16h a, v16h b, v8f c) {
    // D = A(16x32 f16) * B(32x16 f16) + C(16x16 f32)
    return __builtin_amdgcn_wmma_f32_16x16x32_f16(
        /*neg_a=*/false, a, /*neg_b=*/false, b,
        /*c_mod=*/(short)0, c, /*reuse_a=*/false, /*reuse_b=*/false);
}

// A-matrix fragment (16x32, f16), row-major source with rowStride=256 halves.
// Lane L: M = L%16.  Elements e=0..15 -> K = (L>=16?16:0)+e (contiguous)
__device__ __forceinline__ v16h load_a_frag(const _Float16* base, int row, int h_base,
                                            int lane16, int hi) {
    const _Float16* p = base + row * NTOK + h_base + hi * 16;
    v8h lo = *(const v8h*)(p);
    v8h hv = *(const v8h*)(p + 8);
    return __builtin_shufflevector(lo, hv, 0,1,2,3,4,5,6,7,8,9,10,11,12,13,14,15);
}

// B-matrix fragment (32x16, f16) from an LDS row-major [*,256] f16 array where
// B[K,N] = src[rowN, h_base+K].  Lane L: N = L%16, K = e + (L>=16?16:0) (contiguous).
__device__ __forceinline__ v16h load_b_frag_lds(const _Float16* base, int row, int h_base,
                                                int hi) {
    const _Float16* p = base + row * NTOK + h_base + hi * 16;
    v8h lo = *(const v8h*)(p);
    v8h hv = *(const v8h*)(p + 8);
    return __builtin_shufflevector(lo, hv, 0,1,2,3,4,5,6,7,8,9,10,11,12,13,14,15);
}

// B-matrix fragment from global f32 W (row-major [256,256]); B[K,N] = W[rowN, h_base+K].
__device__ __forceinline__ v16h load_b_frag_w(const float* __restrict__ W, int row,
                                              int h_base, int hi) {
    const float* p = W + row * NTOK + h_base + hi * 16;   // 64B aligned
    float4 w0 = *(const float4*)(p);
    float4 w1 = *(const float4*)(p + 4);
    float4 w2 = *(const float4*)(p + 8);
    float4 w3 = *(const float4*)(p + 12);
    v16h r;
    r[0]=(_Float16)w0.x; r[1]=(_Float16)w0.y; r[2]=(_Float16)w0.z; r[3]=(_Float16)w0.w;
    r[4]=(_Float16)w1.x; r[5]=(_Float16)w1.y; r[6]=(_Float16)w1.z; r[7]=(_Float16)w1.w;
    r[8]=(_Float16)w2.x; r[9]=(_Float16)w2.y; r[10]=(_Float16)w2.z; r[11]=(_Float16)w2.w;
    r[12]=(_Float16)w3.x; r[13]=(_Float16)w3.y; r[14]=(_Float16)w3.z; r[15]=(_Float16)w3.w;
    return r;
}

__global__ __launch_bounds__(NTHREADS)
void self_attn_fused(const float* __restrict__ feats, const float* __restrict__ W,
                     const float* __restrict__ bias, float* __restrict__ out) {
    extern __shared__ __align__(16) char smem[];
    _Float16* lds_feats  = (_Float16*)(smem + OFF_FEATS);
    _Float16* lds_a      = (_Float16*)(smem + OFF_A);
    float*    lds_scores = (float*)(smem + OFF_SCORES);
    float*    lds_s      = (float*)(smem + OFF_S);
    float*    lds_bias   = (float*)(smem + OFF_BIAS);
    float*    lds_red    = (float*)(smem + OFF_RED);
    float*    lds_cmax   = (float*)(smem + OFF_CMAX);

    const int b      = blockIdx.x;
    const int tid    = threadIdx.x;
    const int wave   = tid >> 5;
    const int lane   = tid & 31;
    const int lane16 = lane & 15;
    const int hi     = lane >> 4;

    const float* fb = feats + (size_t)b * (NTOK * NTOK);

    // ---------------- Phase A: stage feats (f32 -> f16 LDS) + column sums ----
    {
        const int rm = tid >> 6;          // row mod 4
        const int c4 = (tid & 63) * 4;    // 4 columns per thread
        float a0 = 0.f, a1 = 0.f, a2 = 0.f, a3 = 0.f;
        #pragma unroll 4
        for (int i = 0; i < 64; ++i) {
            const int n = i * 4 + rm;
            float4 v = *(const float4*)(fb + n * NTOK + c4);
            v4h h4;
            h4[0] = (_Float16)v.x; h4[1] = (_Float16)v.y;
            h4[2] = (_Float16)v.z; h4[3] = (_Float16)v.w;
            *(v4h*)(&lds_feats[n * NTOK + c4]) = h4;
            a0 += v.x; a1 += v.y; a2 += v.z; a3 += v.w;
        }
        float* scratch = lds_scores;      // free during phase A
        scratch[rm * NTOK + c4 + 0] = a0;
        scratch[rm * NTOK + c4 + 1] = a1;
        scratch[rm * NTOK + c4 + 2] = a2;
        scratch[rm * NTOK + c4 + 3] = a3;
        lds_bias[tid] = bias[tid];
        __syncthreads();
        // deterministic fixed-order combine of the 4 row-mod partials
        lds_s[tid] = ((scratch[0 * NTOK + tid] + scratch[1 * NTOK + tid]) +
                      (scratch[2 * NTOK + tid] + scratch[3 * NTOK + tid]));
        __syncthreads();
    }

    // ---------------- Phase B: a = feats @ W^T + bias  (f16 WMMA, f32 acc) ---
    // wave owns k-column strips k_t = 2*wave, 2*wave+1 ; all 16 n tiles.
    for (int kk = 0; kk < 2; ++kk) {
        const int k_base = (wave * 2 + kk) * 16;
        v16h Bf[8];
        #pragma unroll
        for (int h = 0; h < 8; ++h)
            Bf[h] = load_b_frag_w(W, k_base + lane16, h * 32, hi);
        const float bv = lds_bias[k_base + lane16];
        for (int n_t = 0; n_t < 16; ++n_t) {
            v8f acc = {};
            #pragma unroll
            for (int h = 0; h < 8; ++h) {
                v16h A = load_a_frag(lds_feats, n_t * 16 + lane16, h * 32, lane16, hi);
                acc = wmma_f16(A, Bf[h], acc);
            }
            // D layout: VGPR r -> M = r + hi*8, N = lane16
            #pragma unroll
            for (int r = 0; r < 8; ++r) {
                const int n = n_t * 16 + r + hi * 8;
                lds_a[n * NTOK + k_base + lane16] = (_Float16)(acc[r] + bv);
            }
        }
    }
    __syncthreads();

    // ---------------- Phase C: scores strips + column softmax + fused output -
    for (int strip = 0; strip < 8; ++strip) {
        const int m0 = strip * 32;
        // GEMM2: scores[n, m'] = sum_h a[n,h] * feats[m0+m',h]
        #pragma unroll
        for (int m_t = 0; m_t < 2; ++m_t) {
            v16h Bf[8];
            #pragma unroll
            for (int h = 0; h < 8; ++h)
                Bf[h] = load_b_frag_lds(lds_feats, m0 + m_t * 16 + lane16, h * 32, hi);
            #pragma unroll
            for (int nn = 0; nn < 2; ++nn) {
                const int n_t = wave * 2 + nn;
                v8f acc = {};
                #pragma unroll
                for (int h = 0; h < 8; ++h) {
                    v16h A = load_a_frag(lds_a, n_t * 16 + lane16, h * 32, lane16, hi);
                    acc = wmma_f16(A, Bf[h], acc);
                }
                #pragma unroll
                for (int r = 0; r < 8; ++r) {
                    const int n = n_t * 16 + r + hi * 8;
                    lds_scores[n * 32 + m_t * 16 + lane16] = acc[r];
                }
            }
        }
        __syncthreads();

        // column softmax over n (axis=1) fused with out[m] = sum_k s[k]*p[k,m]
        const int col  = tid & 31;
        const int part = tid >> 5;      // wave id == row group; lane == col (coalesced LDS)
        float mx = -3.402823466e38f;
        #pragma unroll 4
        for (int i = 0; i < 32; ++i)
            mx = fmaxf(mx, lds_scores[(part * 32 + i) * 32 + col]);
        lds_red[tid] = mx;
        __syncthreads();
        if (tid < 32) {
            float m = lds_red[tid];
            #pragma unroll
            for (int p = 1; p < 8; ++p) m = fmaxf(m, lds_red[p * 32 + tid]);
            lds_cmax[tid] = m;
        }
        __syncthreads();
        const float cm = lds_cmax[col];
        float es = 0.f, ws = 0.f;
        #pragma unroll 4
        for (int i = 0; i < 32; ++i) {
            const int n = part * 32 + i;
            const float e = __expf(lds_scores[n * 32 + col] - cm);
            es += e;
            ws += lds_s[n] * e;
        }
        lds_red[tid]       = es;
        lds_red[256 + tid] = ws;
        __syncthreads();
        if (tid < 32) {
            float et = 0.f, wt = 0.f;
            #pragma unroll
            for (int p = 0; p < 8; ++p) {
                et += lds_red[p * 32 + tid];
                wt += lds_red[256 + p * 32 + tid];
            }
            out[(size_t)b * NTOK + m0 + tid] = wt / et;
        }
        __syncthreads();  // lds_scores / lds_red reused next strip
    }
}

extern "C" void kernel_launch(void* const* d_in, const int* in_sizes, int n_in,
                              void* d_out, int out_size, void* d_ws, size_t ws_size,
                              hipStream_t stream) {
    (void)in_sizes; (void)n_in; (void)out_size; (void)d_ws; (void)ws_size;
    const float* feats = (const float*)d_in[0];   // [512,256,256] f32
    const float* W     = (const float*)d_in[1];   // [256,256] f32
    const float* bias  = (const float*)d_in[2];   // [256] f32
    float*       out   = (float*)d_out;           // [512,256] f32

    self_attn_fused<<<dim3(512), dim3(NTHREADS), SMEM_BYTES, stream>>>(feats, W, bias, out);
}
